// LearnedDistance_29411936043390
// MI455X (gfx1250) — matrix operand order repaired and checked by reading, not compiled, not run on hardware
//
#include <hip/hip_runtime.h>
#include <hip/hip_bf16.h>

// ---------------------------------------------------------------------------
// MI455X (gfx1250) fused implementation of the "LearnedDistance" reference.
//   N=2048, DH=256, DK=64, FF=32
// prep GEMMs (q,k f16; v f16 transposed) with WMMA, then one fused kernel per
// 16-row i-tile: scores -> edge-MLP -> softmax -> P@V, logits LDS-resident,
// all matrix math on v_wmma_f32_16x16x32_f16.
// ---------------------------------------------------------------------------

typedef _Float16 v16h __attribute__((ext_vector_type(16)));
typedef _Float16 v8h  __attribute__((ext_vector_type(8)));
typedef _Float16 v4h  __attribute__((ext_vector_type(4)));
typedef float    v8f  __attribute__((ext_vector_type(8)));

#define NN 2048
#define DH 256
#define DK 64
#define FF 32

static __device__ __forceinline__ v8f wmma16(v16h a, v16h b, v8f c) {
  // D = A(16x32 f16) * B(32x16 f16) + C(16x16 f32)
  return __builtin_amdgcn_wmma_f32_16x16x32_f16(false, a, false, b,
                                                (short)0, c, false, false);
}

static __device__ __forceinline__ void lds_fence() {
  // Same-wave LDS transpose round-trip: DS ops are in-order per wave in HW;
  // the wait + memory clobber also stops the compiler from reordering.
  asm volatile("s_wait_dscnt 0x0" ::: "memory");
}

// ---------------------------------------------------------------------------
// Prep: Y(f16) = X(f32)[2048,K] @ W(f32)[K,NC] + bias, one wave per 16x16
// tile. All dims compile-time so addressing is constant-folded and the
// transpose variant has straight-line stores.
// ---------------------------------------------------------------------------
template <int K, int NC, bool TR>
__global__ __launch_bounds__(256) void prep_gemm_f16(
    const float* __restrict__ X, const float* __restrict__ W,
    const float* __restrict__ bias, _Float16* __restrict__ Y)
{
  constexpr int NTN = NC / 16;
  const int lane = threadIdx.x & 31;
  const int wid  = (int)((blockIdx.x * blockDim.x + threadIdx.x) >> 5);
  const int i0 = (wid / NTN) << 4;
  const int n0 = (wid % NTN) << 4;
  const int n  = lane & 15;
  const int hi = lane >> 4;

  const float bb = bias[n0 + n];
  v8f c;
#pragma unroll
  for (int r = 0; r < 8; ++r) c[r] = bb;   // bias pre-loaded into C operand

#pragma unroll
  for (int kk0 = 0; kk0 < K; kk0 += 32) {
    // A operand (16x32 f16): row = lane%16, K = kk0 + 16*(lane/16) + 0..15.
    const float* xr = X + (size_t)(i0 + n) * K + kk0 + (hi ? 16 : 0);
    v16h a;
#pragma unroll
    for (int t = 0; t < 8; ++t) { a[t] = (_Float16)xr[t]; a[8 + t] = (_Float16)xr[8 + t]; }
    // B operand (32x16 f16): col = n0+n, rows kk0 + (hi?16:0) .. +15.
    const float* wr = W + (size_t)(kk0 + (hi ? 16 : 0)) * NC + (n0 + n);
    v16h b;
#pragma unroll
    for (int t = 0; t < 16; ++t) b[t] = (_Float16)wr[t * NC];
    c = wmma16(a, b, c);
  }

#pragma unroll
  for (int r = 0; r < 8; ++r) {
    const int row = i0 + r + 8 * hi;    // C layout: M = r + 8*(lane/16)
    const int col = n0 + n;
    if (TR) Y[(size_t)col * NN + row] = (_Float16)c[r];
    else    Y[(size_t)row * NC + col] = (_Float16)c[r];
  }
}

// ---------------------------------------------------------------------------
// Fused: one block = 16 rows of i. 8 waves. LDS holds logits[16][2048] (f32)
// plus per-wave transpose scratch. Phases: A (scores+MLP) -> B (softmax) ->
// C (feature = focus @ v via WMMA).
// ---------------------------------------------------------------------------
__global__ __launch_bounds__(256) void fused_attn(
    const _Float16* __restrict__ qh,   // [N][DK]
    const _Float16* __restrict__ kh,   // [N][DK]
    const _Float16* __restrict__ vth,  // [DH][N] (transposed)
    const float* __restrict__ adj, const float* __restrict__ dense,
    const float* __restrict__ W1, const float* __restrict__ b1,
    const float* __restrict__ W2, const float* __restrict__ b2,
    const float* __restrict__ W3, const float* __restrict__ b3,
    float* __restrict__ out)
{
  extern __shared__ char smem[];
  float*    logits = (float*)smem;                           // 16*2048*4 = 128KB
  float*    red    = (float*)(smem + 16 * NN * 4);           // 16*16 f32
  _Float16* wsc    = (_Float16*)(smem + 16 * NN * 4 + 1024); // 8 waves * 1536 halfs

  const int tid  = threadIdx.x;
  const int lane = tid & 31;
  const int w    = tid >> 5;         // wave id 0..7
  const int n    = lane & 15;
  const int hi   = lane >> 4;
  const int i0   = blockIdx.x << 4;

  _Float16* feats = wsc + w * 1536;  // [256 elems][4] halfs (2KB / wave)
  _Float16* h1b   = feats + 1024;    // [16][32] halfs (1KB / wave)

  // ---- wave-invariant operands -------------------------------------------
  v16h ak0, ak1;                     // A = k_i (16x64), split into two K=32 chunks
  {
    const _Float16* kr = kh + (size_t)(i0 + n) * DK + (hi ? 16 : 0);
#pragma unroll
    for (int t = 0; t < 8; ++t) {
      ak0[t] = kr[t];      ak0[8 + t] = kr[8 + t];
      ak1[t] = kr[32 + t]; ak1[8 + t] = kr[40 + t];
    }
  }
  v16h w1b0, w1b1, w2b0, w2b1;       // B operands: W1 zero-padded K 3->32, W2 K=32
  {
#pragma unroll
    for (int t = 0; t < 16; ++t) {
      const int kk = (hi ? 16 : 0) + t;
      const bool live = (!hi && kk < 3);
      w1b0[t] = live ? (_Float16)W1[kk * FF + n]      : (_Float16)0.f;
      w1b1[t] = live ? (_Float16)W1[kk * FF + 16 + n] : (_Float16)0.f;
      w2b0[t] = (_Float16)W2[kk * FF + n];
      w2b1[t] = (_Float16)W2[kk * FF + 16 + n];
    }
  }
  const float w3a = W3[n], w3b = W3[16 + n];
  const float b1a = b1[n], b1bv = b1[16 + n];
  const float b2a = b2[n], b2bv = b2[16 + n];
  const float b3v = b3[0];

  // ---- Phase A: logits[i_local][j] for all j, waves stride over j-tiles ---
  for (int jt = w; jt < NN / 16; jt += 8) {
    const int j0 = jt << 4;
    // prefetch next tile's adj/dense rows into the cache hierarchy
    if (jt + 8 < NN / 16) {
      const size_t pidx = (size_t)(i0 + hi * 8) * NN + ((jt + 8) << 4) + n;
      __builtin_prefetch(adj + pidx, 0, 0);
      __builtin_prefetch(dense + pidx, 0, 0);
    }
    // B = q_j (64x16), two K=32 chunks; contiguous 32B per lane
    v16h bq0, bq1;
    {
      const _Float16* qr = qh + (size_t)(j0 + n) * DK + (hi ? 16 : 0);
#pragma unroll
      for (int t = 0; t < 16; ++t) { bq0[t] = qr[t]; bq1[t] = qr[32 + t]; }
    }
    v8f s;
#pragma unroll
    for (int r = 0; r < 8; ++r) s[r] = 0.f;
    s = wmma16(ak0, bq0, s);
    s = wmma16(ak1, bq1, s);           // S[m,n] = <k_{i0+m}, q_{j0+n}>

    // feats = [score, adj, dense, 0] staged to LDS (C-layout -> row-major)
#pragma unroll
    for (int r = 0; r < 8; ++r) {
      const int m = r + 8 * hi;                       // i_local
      const size_t idx = (size_t)(i0 + m) * NN + j0 + n;
      v4h f;
      f[0] = (_Float16)s[r];
      f[1] = (_Float16)adj[idx];
      f[2] = (_Float16)dense[idx];
      f[3] = (_Float16)0.f;
      *(v4h*)(feats + (size_t)(m * 16 + n) * 4) = f;
    }
    lds_fence();

    // 16 groups: group g = i_local, rows inside group = j_local
    for (int g = 0; g < 16; ++g) {
      v16h af;
#pragma unroll
      for (int t = 0; t < 16; ++t) af[t] = (_Float16)0.f;
      if (!hi) {                                       // only K 0..2 nonzero
        v4h f = *(v4h*)(feats + (size_t)(g * 16 + n) * 4);
        af[0] = f[0]; af[1] = f[1]; af[2] = f[2]; af[3] = f[3];
      }
      v8f c1a, c1b;
#pragma unroll
      for (int r = 0; r < 8; ++r) { c1a[r] = b1a; c1b[r] = b1bv; }
      c1a = wmma16(af, w1b0, c1a);                     // layer 1 (3->32, padded)
      c1b = wmma16(af, w1b1, c1b);

#pragma unroll
      for (int r = 0; r < 8; ++r) {                    // relu, f16, stage to LDS
        const int m = r + 8 * hi;
        h1b[m * FF + n]      = (_Float16)fmaxf(c1a[r], 0.f);
        h1b[m * FF + 16 + n] = (_Float16)fmaxf(c1b[r], 0.f);
      }
      lds_fence();

      v16h ah;                                         // h1 in A-layout
      {
        const _Float16* hr = h1b + n * FF + (hi ? 16 : 0);
        v8h x0 = *(const v8h*)hr;
        v8h x1 = *(const v8h*)(hr + 8);
#pragma unroll
        for (int t = 0; t < 8; ++t) { ah[t] = x0[t]; ah[8 + t] = x1[t]; }
      }
      v8f c2a, c2b;
#pragma unroll
      for (int r = 0; r < 8; ++r) { c2a[r] = b2a; c2b[r] = b2bv; }
      c2a = wmma16(ah, w2b0, c2a);                     // layer 2 (32->32)
      c2b = wmma16(ah, w2b1, c2b);

      // layer 3 (32->1) on VALU: relu, dot with W3, lane-reduce within 16
#pragma unroll
      for (int r = 0; r < 8; ++r) {
        float p = fmaxf(c2a[r], 0.f) * w3a + fmaxf(c2b[r], 0.f) * w3b;
        p += __shfl_xor(p, 1);
        p += __shfl_xor(p, 2);
        p += __shfl_xor(p, 4);
        p += __shfl_xor(p, 8);
        if (n == 0) logits[g * NN + j0 + r + 8 * hi] = p + b3v;
      }
    }
  }
  __syncthreads();

  // ---- Phase B: row softmax over j (16 rows x 2048), all in LDS -----------
  {
    const int row = tid >> 4;          // 0..15
    const int seg = tid & 15;          // 16 threads per row, 128 j's each
    float* Lrow = logits + row * NN;
    const int jb = seg * 128;
    float lm = -3.4e38f;
    for (int j = jb; j < jb + 128; ++j) lm = fmaxf(lm, Lrow[j]);
    red[row * 16 + seg] = lm;
    __syncthreads();
    float mrow = -3.4e38f;
#pragma unroll
    for (int t = 0; t < 16; ++t) mrow = fmaxf(mrow, red[row * 16 + t]);
    __syncthreads();
    float ls = 0.f;
    for (int j = jb; j < jb + 128; ++j) ls += __expf(Lrow[j] - mrow);
    red[row * 16 + seg] = ls;
    __syncthreads();
    float lsum = 0.f;
#pragma unroll
    for (int t = 0; t < 16; ++t) lsum += red[row * 16 + t];
    const float inv = 1.f / lsum;
    for (int j = jb; j < jb + 128; ++j) Lrow[j] = __expf(Lrow[j] - mrow) * inv;
  }
  __syncthreads();

  // ---- Phase C: feature(16 x 256) = focus(16 x 2048) @ v ------------------
  {
    const int d0a = w << 4;            // this wave's two 16-col slices of DH
    const int d0b = (w + 8) << 4;
    v8f acc0, acc1;
#pragma unroll
    for (int r = 0; r < 8; ++r) { acc0[r] = 0.f; acc1[r] = 0.f; }

    for (int jc = 0; jc < NN / 32; ++jc) {
      v16h ap;                          // A = P chunk: rows i_local, K = j
      {
        const float* pr = logits + n * NN + jc * 32 + (hi ? 16 : 0);
#pragma unroll
        for (int t = 0; t < 8; ++t) { ap[t] = (_Float16)pr[t]; ap[8 + t] = (_Float16)pr[8 + t]; }
      }
      v16h bv0, bv1;                    // B = vT tiles (contiguous 32B per lane)
      {
        const _Float16* v0 = vth + (size_t)(d0a + n) * NN + jc * 32 + (hi ? 16 : 0);
        const _Float16* v1 = vth + (size_t)(d0b + n) * NN + jc * 32 + (hi ? 16 : 0);
#pragma unroll
        for (int t = 0; t < 16; ++t) { bv0[t] = v0[t]; bv1[t] = v1[t]; }
      }
      acc0 = wmma16(ap, bv0, acc0);
      acc1 = wmma16(ap, bv1, acc1);
    }
#pragma unroll
    for (int r = 0; r < 8; ++r) {
      const int row = i0 + r + 8 * hi;
      out[(size_t)row * DH + d0a + n] = acc0[r];
      out[(size_t)row * DH + d0b + n] = acc1[r];
    }
  }
}

// ---------------------------------------------------------------------------
extern "C" void kernel_launch(void* const* d_in, const int* in_sizes, int n_in,
                              void* d_out, int out_size, void* d_ws, size_t ws_size,
                              hipStream_t stream) {
  (void)in_sizes; (void)n_in; (void)out_size; (void)ws_size;
  const float* x     = (const float*)d_in[0];
  const float* adj   = (const float*)d_in[1];
  const float* dense = (const float*)d_in[2];
  const float* Wq    = (const float*)d_in[3];
  const float* bq    = (const float*)d_in[4];
  const float* Wk    = (const float*)d_in[5];
  const float* bk    = (const float*)d_in[6];
  const float* Wv    = (const float*)d_in[7];
  const float* bv    = (const float*)d_in[8];
  const float* W1    = (const float*)d_in[9];
  const float* b1    = (const float*)d_in[10];
  const float* W2    = (const float*)d_in[11];
  const float* b2    = (const float*)d_in[12];
  const float* W3    = (const float*)d_in[13];
  const float* b3    = (const float*)d_in[14];

  _Float16* qh  = (_Float16*)d_ws;           // [2048][64]
  _Float16* kh  = qh + (size_t)NN * DK;      // [2048][64]
  _Float16* vth = kh + (size_t)NN * DK;      // [256][2048]

  // prep GEMMs: one wave per 16x16 output tile, all dims compile-time
  constexpr int tiles_qk = (NN / 16) * (DK / 16);   // 512 waves
  constexpr int tiles_v  = (NN / 16) * (DH / 16);   // 2048 waves
  prep_gemm_f16<DH, DK, false><<<tiles_qk * 32 / 256, 256, 0, stream>>>(x, Wq, bq, qh);
  prep_gemm_f16<DH, DK, false><<<tiles_qk * 32 / 256, 256, 0, stream>>>(x, Wk, bk, kh);
  prep_gemm_f16<DH, DH, true ><<<tiles_v  * 32 / 256, 256, 0, stream>>>(x, Wv, bv, vth);

  // fused kernel: 128 i-tiles, 8 waves each, 153KB dynamic LDS
  const size_t smem = (size_t)16 * NN * 4 + 1024 + (size_t)8 * 1536 * 2;
  fused_attn<<<NN / 16, 256, smem, stream>>>(qh, kh, vth, adj, dense,
                                             W1, b1, W2, b2, W3, b3,
                                             (float*)d_out);
}
